// SeqIVModel_81415400063226
// MI455X (gfx1250) — matrix operand order repaired and checked by reading, not compiled, not run on hardware
//
#include <hip/hip_runtime.h>
#include <cstddef>

// ---------------------------------------------------------------------------
// CDNA5 (gfx1250) persistent bidirectional-LSTM decoder.
//   - one workgroup per 16-row batch tile (WMMA M=16), whole T-loop in-kernel
//   - all gate / proj / head GEMMs via v_wmma_f32_16x16x32_bf16
//   - weights pre-packed into WMMA B-fragment order (bf16) in d_ws
// ---------------------------------------------------------------------------

typedef __attribute__((ext_vector_type(16))) __bf16 bvec16;
typedef __attribute__((ext_vector_type(8)))  __bf16 bvec8;
typedef __attribute__((ext_vector_type(8)))  float  fvec8;

#define NTHR   256          // 8 waves
#define B_TOT  512
#define HDIM   128
#define GDIM   512          // 4*H gates
#define TSEQ   128
#define BT     16           // batch tile (= WMMA M)

// packed-weight offsets in d_ws (bf16 elements)
#define OFF_WHH0  0                         // (2*512, 128)  -> 131072
#define OFF_WIH1  131072                    // (2*512, 256)  -> 262144
#define OFF_WHH1  393216                    // (2*512, 128)  -> 131072
#define OFF_PROJ  524288                    // (128, 256)    ->  32768
#define OFF_HEAD1 557056                    // (128, 128)    ->  16384

// dynamic-LDS layout (bytes, all 16B aligned)
#define SM_GATES  0         // float [16][512]          32768
#define SM_C      32768     // float [4][16][128]       32768
#define SM_H1F    65536     // float [16][256]          16384
#define SM_H0CAT  81920     // bf16  [16][256]           8192
#define SM_H1CAT  90112     // bf16  [16][256]           8192
#define SM_LN     98304     // bf16  [16][256]           8192
#define SM_PROJ   106496    // bf16  [16][128]           4096
#define SM_RELU   110592    // bf16  [16][128]           4096
#define SM_BSUM   114688    // float [4][512]            8192
#define SM_WIH0   122880    // float [2][512]            4096
#define SM_MISC   126976    // float xv/prev/mu/rv [64]   256
#define SM_TOTAL  127232

__device__ __forceinline__ float sigm(float x) { return 1.0f / (1.0f + expf(-x)); }

// A-fragment (16x32 bf16) from LDS, ISA 16-bit A layout:
//   lanes 0-15 : halves 0..7 = K k0..k0+7,  halves 8..15 = K k0+16..k0+23
//   lanes 16-31: same with +8 on K
__device__ __forceinline__ bvec16 load_a_frag(const __bf16* base, int rowstride,
                                              int k0, int lane) {
  const __bf16* p = base + (lane & 15) * rowstride + k0 + ((lane >> 4) << 3);
  bvec8 lo = *(const bvec8*)p;
  bvec8 hi = *(const bvec8*)(p + 16);
  bvec16 a;
#pragma unroll
  for (int i = 0; i < 8; ++i) { a[i] = lo[i]; a[i + 8] = hi[i]; }
  return a;
}

// B-fragment (32x16 bf16) from pre-packed global memory: each lane's 16
// halves are contiguous (32 bytes) -> two global_load_b128.
__device__ __forceinline__ bvec16 load_b_frag(const __bf16* wbase, int tile, int lane) {
  return *(const bvec16*)(wbase + ((size_t)tile << 9) + lane * 16);
}

__device__ __forceinline__ fvec8 wmma_bf16(bvec16 a, bvec16 b, fvec8 c) {
  return __builtin_amdgcn_wmma_f32_16x16x32_bf16(false, a, false, b, (short)0, c,
                                                 false, false);
}

// ---------------------------------------------------------------------------
// pre-pass: repack a row-major (nr x kc) f32 weight matrix (gates = x @ W^T,
// so B(k,n) = W[n][k]) into bf16 WMMA-B fragment order.
// packed[((tn*kt + tk)*32 + lane)*16 + i] = W[tn*16 + (lane&15)][tk*32 + (lane>>4)*8 + (i&7) + (i>>3)*16]
// ---------------------------------------------------------------------------
__global__ void pack_wmma_b_kernel(const float* __restrict__ W, __bf16* __restrict__ dst,
                                   int nr, int kc) {
  int idx = blockIdx.x * blockDim.x + threadIdx.x;
  if (idx >= nr * kc) return;
  int kt   = kc >> 5;
  int i    = idx & 15;
  int lane = (idx >> 4) & 31;
  int tile = idx >> 9;
  int tk   = tile % kt;
  int tn   = tile / kt;
  int n    = tn * 16 + (lane & 15);
  int k    = tk * 32 + ((lane >> 4) << 3) + (i & 7) + ((i >> 3) << 4);
  dst[idx] = (__bf16)W[(size_t)n * kc + k];
}

// ---------------------------------------------------------------------------
// main persistent kernel: grid = 32 blocks x 256 threads
// ---------------------------------------------------------------------------
__global__ void __launch_bounds__(NTHR)
seq_iv_lstm_kernel(const float* __restrict__ physical,   // (512,16)
                   const float* __restrict__ target_seq, // (512,128)
                   const float* __restrict__ enc_W1,     // (256,16)
                   const float* __restrict__ enc_b1,
                   const float* __restrict__ enc_ln1_g,
                   const float* __restrict__ enc_ln1_b,
                   const float* __restrict__ enc_W2,     // (128,256)
                   const float* __restrict__ enc_b2,
                   const float* __restrict__ enc_ln2_g,
                   const float* __restrict__ enc_ln2_b,
                   const float* __restrict__ Wih0,       // (2,512,1)
                   const float* __restrict__ bih0,       // (2,512)
                   const float* __restrict__ bhh0,
                   const float* __restrict__ bih1,
                   const float* __restrict__ bhh1,
                   const float* __restrict__ ln_g,       // (256)
                   const float* __restrict__ ln_b,
                   const float* __restrict__ proj_b,     // (128)
                   const float* __restrict__ head_b1,    // (128)
                   const float* __restrict__ head_W2,    // (128)
                   const float* __restrict__ head_b2,    // (1)
                   const float* __restrict__ init_input, // (1)
                   const __bf16* __restrict__ ws,        // packed weights
                   float* __restrict__ out)              // (512,128)
{
  const int tid  = threadIdx.x;
  const int lane = tid & 31;
  const int wave = tid >> 5;
  const int b0   = blockIdx.x * BT;

  extern __shared__ __align__(16) char smem[];
  float*  gates  = (float*)(smem + SM_GATES);   // scratch: gates / encoder z
  float*  cst    = (float*)(smem + SM_C);       // [cell][b][j]
  float*  h1f32  = (float*)(smem + SM_H1F);     // f32 copy of concat(h1f,h1b)
  __bf16* h0cat  = (__bf16*)(smem + SM_H0CAT);  // [16][256] cols 0-127 fwd, 128-255 bwd
  __bf16* h1cat  = (__bf16*)(smem + SM_H1CAT);
  __bf16* lnbuf  = (__bf16*)(smem + SM_LN);
  __bf16* projb  = (__bf16*)(smem + SM_PROJ);
  __bf16* relub  = (__bf16*)(smem + SM_RELU);
  float*  bsum   = (float*)(smem + SM_BSUM);    // bih+bhh per cell
  float*  wih0s  = (float*)(smem + SM_WIH0);
  float*  xv       = (float*)(smem + SM_MISC);
  float*  predprev = xv + 16;
  float*  muv      = predprev + 16;
  float*  rvv      = muv + 16;

  const __bf16* wsWhh0  = ws + OFF_WHH0;
  const __bf16* wsWih1  = ws + OFF_WIH1;
  const __bf16* wsWhh1  = ws + OFF_WHH1;
  const __bf16* wsProj  = ws + OFF_PROJ;
  const __bf16* wsHead1 = ws + OFF_HEAD1;

  // ---- one-time init: fused biases, Wih0 copy, c=0, prev pred=0 ------------
  for (int i = tid; i < 4 * GDIM; i += NTHR) {
    int cell = i >> 9, n = i & 511;
    bsum[i] = (cell < 2) ? (bih0[cell * 512 + n] + bhh0[cell * 512 + n])
                         : (bih1[(cell - 2) * 512 + n] + bhh1[(cell - 2) * 512 + n]);
  }
  for (int i = tid; i < 2 * GDIM; i += NTHR) wih0s[i] = Wih0[i];
  for (int i = tid; i < 4 * BT * HDIM; i += NTHR) cst[i] = 0.0f;
  if (tid < BT) predprev[tid] = 0.0f;
  __syncthreads();

  // ---- encoder: h0 = tanh(LN(relu(LN(phys@W1^T+b1))@W2^T+b2)) --------------
  for (int idx = tid; idx < BT * 256; idx += NTHR) {        // z1 -> gates
    int b = idx >> 8, j = idx & 255;
    const float* pr = physical + (size_t)(b0 + b) * 16;
    const float* w  = enc_W1 + j * 16;
    float s = enc_b1[j];
#pragma unroll
    for (int k = 0; k < 16; ++k) s += pr[k] * w[k];
    gates[idx] = s;
  }
  __syncthreads();
  if (tid < BT) {
    const float* row = gates + tid * 256;
    float m = 0.f; for (int j = 0; j < 256; ++j) m += row[j];
    m *= (1.0f / 256.0f);
    float v = 0.f; for (int j = 0; j < 256; ++j) { float d = row[j] - m; v += d * d; }
    muv[tid] = m; rvv[tid] = rsqrtf(v * (1.0f / 256.0f) + 1e-5f);
  }
  __syncthreads();
  for (int idx = tid; idx < BT * 256; idx += NTHR) {        // e -> h1f32
    int b = idx >> 8, j = idx & 255;
    float e = (gates[idx] - muv[b]) * rvv[b] * enc_ln1_g[j] + enc_ln1_b[j];
    h1f32[idx] = fmaxf(e, 0.0f);
  }
  __syncthreads();
  for (int idx = tid; idx < BT * HDIM; idx += NTHR) {       // z2 -> gates
    int b = idx >> 7, j = idx & 127;
    const float* e = h1f32 + b * 256;
    const float* w = enc_W2 + j * 256;
    float s = enc_b2[j];
    for (int k = 0; k < 256; ++k) s += e[k] * w[k];
    gates[idx] = s;
  }
  __syncthreads();
  if (tid < BT) {
    const float* row = gates + tid * 128;
    float m = 0.f; for (int j = 0; j < 128; ++j) m += row[j];
    m *= (1.0f / 128.0f);
    float v = 0.f; for (int j = 0; j < 128; ++j) { float d = row[j] - m; v += d * d; }
    muv[tid] = m; rvv[tid] = rsqrtf(v * (1.0f / 128.0f) + 1e-5f);
  }
  __syncthreads();
  for (int idx = tid; idx < BT * HDIM; idx += NTHR) {       // h0 broadcast
    int b = idx >> 7, j = idx & 127;
    float h0v = tanhf((gates[idx] - muv[b]) * rvv[b] * enc_ln2_g[j] + enc_ln2_b[j]);
    __bf16 hb = (__bf16)h0v;
    h0cat[b * 256 + j] = hb;       h0cat[b * 256 + 128 + j] = hb;
    h1cat[b * 256 + j] = hb;       h1cat[b * 256 + 128 + j] = hb;
    h1f32[b * 256 + j] = h0v;      h1f32[b * 256 + 128 + j] = h0v;
  }
  __syncthreads();

  // ---- recurrent time loop -------------------------------------------------
#pragma unroll 1
  for (int t = 0; t < TSEQ; ++t) {
    if (tid < BT)
      xv[tid] = (t == 0) ? init_input[0]
                         : target_seq[(size_t)(b0 + tid) * TSEQ + (t - 1)];
    __syncthreads();

    // ===== layer 0, directions serialized (gates buffer reused) =====
#pragma unroll 1
    for (int d = 0; d < 2; ++d) {
      const __bf16* Wp = wsWhh0 + (size_t)d * GDIM * HDIM;
      bvec16 a[4];
#pragma unroll
      for (int kt = 0; kt < 4; ++kt)
        a[kt] = load_a_frag(h0cat + d * 128, 256, kt * 32, lane);
#pragma unroll 1
      for (int nt = wave; nt < 32; nt += 8) {
        fvec8 acc = {0.f, 0.f, 0.f, 0.f, 0.f, 0.f, 0.f, 0.f};
#pragma unroll
        for (int kt = 0; kt < 4; ++kt)
          acc = wmma_bf16(a[kt], load_b_frag(Wp, nt * 4 + kt, lane), acc);
        int col = nt * 16 + (lane & 15);
        int rb  = (lane >> 4) << 3;
#pragma unroll
        for (int v = 0; v < 8; ++v) gates[(rb + v) * GDIM + col] = acc[v];
      }
      __syncthreads();
      // elementwise LSTM update, cell = d
      for (int idx = tid; idx < BT * HDIM; idx += NTHR) {
        int b = idx >> 7, j = idx & 127;
        float x = xv[b];
        const float* g  = gates + b * GDIM;
        const float* bs = bsum  + d * GDIM;
        const float* wx = wih0s + d * GDIM;
        float gi = g[j]       + x * wx[j]       + bs[j];
        float gf = g[128 + j] + x * wx[128 + j] + bs[128 + j];
        float gg = g[256 + j] + x * wx[256 + j] + bs[256 + j];
        float go = g[384 + j] + x * wx[384 + j] + bs[384 + j];
        float* cp = cst + ((size_t)d * BT + b) * HDIM + j;
        float cn = sigm(gf) * cp[0] + sigm(gi) * tanhf(gg);
        float hn = sigm(go) * tanhf(cn);
        cp[0] = cn;
        h0cat[b * 256 + d * 128 + j] = (__bf16)hn;
      }
      __syncthreads();
    }

    // ===== layer 1 =====
#pragma unroll 1
    for (int d = 0; d < 2; ++d) {
      const __bf16* Wx = wsWih1 + (size_t)d * GDIM * 256;
      const __bf16* Wh = wsWhh1 + (size_t)d * GDIM * HDIM;
      bvec16 ax[8], ah[4];
#pragma unroll
      for (int kt = 0; kt < 8; ++kt) ax[kt] = load_a_frag(h0cat, 256, kt * 32, lane);
#pragma unroll
      for (int kt = 0; kt < 4; ++kt)
        ah[kt] = load_a_frag(h1cat + d * 128, 256, kt * 32, lane);
#pragma unroll 1
      for (int nt = wave; nt < 32; nt += 8) {
        fvec8 acc = {0.f, 0.f, 0.f, 0.f, 0.f, 0.f, 0.f, 0.f};
#pragma unroll
        for (int kt = 0; kt < 8; ++kt)
          acc = wmma_bf16(ax[kt], load_b_frag(Wx, nt * 8 + kt, lane), acc);
#pragma unroll
        for (int kt = 0; kt < 4; ++kt)
          acc = wmma_bf16(ah[kt], load_b_frag(Wh, nt * 4 + kt, lane), acc);
        int col = nt * 16 + (lane & 15);
        int rb  = (lane >> 4) << 3;
#pragma unroll
        for (int v = 0; v < 8; ++v) gates[(rb + v) * GDIM + col] = acc[v];
      }
      __syncthreads();
      for (int idx = tid; idx < BT * HDIM; idx += NTHR) {
        int b = idx >> 7, j = idx & 127;
        const float* g  = gates + b * GDIM;
        const float* bs = bsum + (2 + d) * GDIM;
        float gi = g[j] + bs[j];
        float gf = g[128 + j] + bs[128 + j];
        float gg = g[256 + j] + bs[256 + j];
        float go = g[384 + j] + bs[384 + j];
        float* cp = cst + ((size_t)(2 + d) * BT + b) * HDIM + j;
        float cn = sigm(gf) * cp[0] + sigm(gi) * tanhf(gg);
        float hn = sigm(go) * tanhf(cn);
        cp[0] = cn;
        h1cat[b * 256 + d * 128 + j] = (__bf16)hn;
        h1f32[b * 256 + d * 128 + j] = hn;
      }
      __syncthreads();
    }

    // ===== LayerNorm over concat(h1f,h1b) =====
    if (tid < BT) {
      const float* row = h1f32 + tid * 256;
      float m = 0.f; for (int j = 0; j < 256; ++j) m += row[j];
      m *= (1.0f / 256.0f);
      float v = 0.f; for (int j = 0; j < 256; ++j) { float dd = row[j] - m; v += dd * dd; }
      muv[tid] = m; rvv[tid] = rsqrtf(v * (1.0f / 256.0f) + 1e-5f);
    }
    __syncthreads();
    for (int idx = tid; idx < BT * 256; idx += NTHR) {
      int b = idx >> 8, j = idx & 255;
      lnbuf[idx] = (__bf16)((h1f32[idx] - muv[b]) * rvv[b] * ln_g[j] + ln_b[j]);
    }
    __syncthreads();

    // ===== proj: (16x256)@(256x128) + b, one N-tile per wave =====
    {
      bvec16 a[8];
#pragma unroll
      for (int kt = 0; kt < 8; ++kt) a[kt] = load_a_frag(lnbuf, 256, kt * 32, lane);
      fvec8 acc = {0.f, 0.f, 0.f, 0.f, 0.f, 0.f, 0.f, 0.f};
#pragma unroll
      for (int kt = 0; kt < 8; ++kt)
        acc = wmma_bf16(a[kt], load_b_frag(wsProj, wave * 8 + kt, lane), acc);
      int col = wave * 16 + (lane & 15);
      int rb  = (lane >> 4) << 3;
      float pb = proj_b[col];
#pragma unroll
      for (int v = 0; v < 8; ++v)
        projb[(rb + v) * HDIM + col] = (__bf16)(acc[v] + pb);
    }
    __syncthreads();

    // ===== head layer 1: relu((16x128)@(128x128) + b1) =====
    {
      bvec16 a[4];
#pragma unroll
      for (int kt = 0; kt < 4; ++kt) a[kt] = load_a_frag(projb, 128, kt * 32, lane);
      fvec8 acc = {0.f, 0.f, 0.f, 0.f, 0.f, 0.f, 0.f, 0.f};
#pragma unroll
      for (int kt = 0; kt < 4; ++kt)
        acc = wmma_bf16(a[kt], load_b_frag(wsHead1, wave * 4 + kt, lane), acc);
      int col = wave * 16 + (lane & 15);
      int rb  = (lane >> 4) << 3;
      float hb = head_b1[col];
#pragma unroll
      for (int v = 0; v < 8; ++v)
        relub[(rb + v) * HDIM + col] = (__bf16)fmaxf(acc[v] + hb, 0.0f);
    }
    __syncthreads();

    // ===== head layer 2 (dot 128) + autoregressive blend + store =====
    if (tid < BT) {
      float s = head_b2[0];
      const __bf16* r = relub + tid * HDIM;
      for (int k = 0; k < HDIM; ++k) s += (float)r[k] * head_W2[k];
      float pred = s;
      if (t > 0) pred = 0.8f * pred + 0.2f * predprev[tid];
      predprev[tid] = pred;
      out[(size_t)(b0 + tid) * TSEQ + t] = pred;
    }
    __syncthreads();
  }
}

// ---------------------------------------------------------------------------
extern "C" void kernel_launch(void* const* d_in, const int* in_sizes, int n_in,
                              void* d_out, int out_size, void* d_ws, size_t ws_size,
                              hipStream_t stream) {
  const float* physical   = (const float*)d_in[0];
  const float* target_seq = (const float*)d_in[1];
  const float* enc_W1     = (const float*)d_in[2];
  const float* enc_b1     = (const float*)d_in[3];
  const float* enc_ln1_g  = (const float*)d_in[4];
  const float* enc_ln1_b  = (const float*)d_in[5];
  const float* enc_W2     = (const float*)d_in[6];
  const float* enc_b2     = (const float*)d_in[7];
  const float* enc_ln2_g  = (const float*)d_in[8];
  const float* enc_ln2_b  = (const float*)d_in[9];
  const float* Wih0       = (const float*)d_in[10];
  const float* Whh0       = (const float*)d_in[11];
  const float* bih0       = (const float*)d_in[12];
  const float* bhh0       = (const float*)d_in[13];
  const float* Wih1       = (const float*)d_in[14];
  const float* Whh1       = (const float*)d_in[15];
  const float* bih1       = (const float*)d_in[16];
  const float* bhh1       = (const float*)d_in[17];
  const float* ln_g       = (const float*)d_in[18];
  const float* ln_b       = (const float*)d_in[19];
  const float* proj_W     = (const float*)d_in[20];
  const float* proj_b     = (const float*)d_in[21];
  const float* head_W1    = (const float*)d_in[22];
  const float* head_b1    = (const float*)d_in[23];
  const float* head_W2    = (const float*)d_in[24];
  const float* head_b2    = (const float*)d_in[25];
  const float* init_input = (const float*)d_in[26];

  __bf16* ws  = (__bf16*)d_ws;
  float*  out = (float*)d_out;

  // pre-pack weights into WMMA-B fragment order (bf16)
  auto blocks = [](int n) { return (n + NTHR - 1) / NTHR; };
  pack_wmma_b_kernel<<<blocks(1024 * 128), NTHR, 0, stream>>>(Whh0,   ws + OFF_WHH0,  1024, 128);
  pack_wmma_b_kernel<<<blocks(1024 * 256), NTHR, 0, stream>>>(Wih1,   ws + OFF_WIH1,  1024, 256);
  pack_wmma_b_kernel<<<blocks(1024 * 128), NTHR, 0, stream>>>(Whh1,   ws + OFF_WHH1,  1024, 128);
  pack_wmma_b_kernel<<<blocks(128 * 256),  NTHR, 0, stream>>>(proj_W, ws + OFF_PROJ,  128, 256);
  pack_wmma_b_kernel<<<blocks(128 * 128),  NTHR, 0, stream>>>(head_W1, ws + OFF_HEAD1, 128, 128);

  // persistent recurrent kernel: 32 batch tiles of 16 rows
  seq_iv_lstm_kernel<<<B_TOT / BT, NTHR, SM_TOTAL, stream>>>(
      physical, target_seq, enc_W1, enc_b1, enc_ln1_g, enc_ln1_b, enc_W2, enc_b2,
      enc_ln2_g, enc_ln2_b, Wih0, bih0, bhh0, bih1, bhh1, ln_g, ln_b, proj_b,
      head_b1, head_W2, head_b2, init_input, ws, out);
}
